// SwinTransformerBlock4D_41377714929808
// MI455X (gfx1250) — matrix operand order repaired and checked by reading, not compiled, not run on hardware
//
#include <hip/hip_runtime.h>
#include <hip/hip_bf16.h>

// ===========================================================================
// Swin-4D block for gfx1250 (MI455X). All GEMMs via v_wmma_f32_16x16x32_f16.
// Bandwidth-bound problem (~180MB traffic, ~42 GFLOP) -> f16 WMMA + f32
// softmax/LN, LDS staging for the P matrix in attention. V is stored
// transposed so every WMMA B-fragment is two 16-byte loads per lane.
// Workspace requirement: ~177 MB (see offsets below).
// ===========================================================================

typedef _Float16 h16;
typedef __attribute__((ext_vector_type(16))) _Float16 v16h;
typedef __attribute__((ext_vector_type(8)))  _Float16 v8h;
typedef __attribute__((ext_vector_type(8)))  float    v8f;

#define WMMA(a, b, c) __builtin_amdgcn_wmma_f32_16x16x32_f16( \
    false, (a), false, (b), (short)0, (c), false, false)

// ----- problem constants -----
// B=2, D=H=W=T=16, C=96, NH=4, hd=24 (pad 32), window 4^4 -> N=256 tokens,
// 256 windows/batch -> 512 windows, HIDDEN=384, TOK = 2*16^4 = 131072.
constexpr float SCALE_  = 0.20412414523193154f;   // 24^-0.5
constexpr float EPS_    = 1e-5f;

// ----- workspace layout (bytes); phases reuse regions -----
constexpr size_t OFF_WIN = 0;                         // 512*256*96 h16 = 25,165,824
constexpr size_t OFF_Q   = 25165824ull;               // 512*4*256*32 h16 = 33,554,432
constexpr size_t OFF_K   = OFF_Q + 33554432ull;
constexpr size_t OFF_V   = OFF_K + 33554432ull;       // vT: 512*4*32*256 h16 (same size)
constexpr size_t OFF_AO  = OFF_V + 33554432ull;       // 25,165,824 (ends 150,994,944)
constexpr size_t OFF_X2  = 0;                         // TOK*96 f32 = 50,331,648 (reuses win/q)
constexpr size_t OFF_H   = 50331648ull;               // TOK*96 h16 = 25,165,824
constexpr size_t OFF_H2  = OFF_H + 25165824ull;       // TOK*384 h16 = 100,663,296
constexpr size_t OFF_WT  = OFF_H2 + 100663296ull;     // converted f16 weights (221,184)
// f16-weight sub-offsets (in halves)
constexpr size_t WQKV  = 0;                // 288*96 = 27648
constexpr size_t WPROJ = 27648;            // 96*96  = 9216
constexpr size_t WFC1  = WPROJ + 9216;     // 384*96 = 36864
constexpr size_t WFC2  = WFC1 + 36864;     // 96*384 = 36864

// ===========================================================================
// WMMA fragment loaders (ISA 7.12.2 layouts, wave32)
// ===========================================================================

union V16u { v16h v; v8h h[2]; };

// A fragment: 16x32 f16 tile, row-major source with leading dim ld (halves).
// Pass A already offset to (row0, k0). Lane r<16 holds row r: K = hi*8..hi*8+7
// in halves 0..7 and K = 16+hi*8.. in halves 8..15 (two 16-byte loads).
__device__ inline v16h load_a_frag(const h16* A, int ld, int lane) {
    int r = lane & 15, hi = lane >> 4;
    const h16* p = A + (size_t)r * ld + hi * 8;
    V16u u;
    u.h[0] = *(const v8h*)(p);
    u.h[1] = *(const v8h*)(p + 16);
    return u.v;
}

// B fragment where B(k,n) = W[n][k] (source stored (N,K) row-major).
// Pass W already offset to (n0, k0). Lane holds col n: K = hi*8..hi*8+7
// in halves 0..7 and K = 16+hi*8.. in halves 8..15 (two 16-byte loads).
__device__ inline v16h load_b_frag_rm(const h16* W, int ldw, int lane) {
    int n = lane & 15, hi = lane >> 4;
    const h16* p = W + (size_t)n * ldw + hi * 8;
    V16u u;
    u.h[0] = *(const v8h*)(p);
    u.h[1] = *(const v8h*)(p + 16);
    return u.v;
}

// ===========================================================================
// helpers
// ===========================================================================

__device__ inline float wave32_sum(float v) {
#pragma unroll
    for (int m = 1; m < 32; m <<= 1) v += __shfl_xor(v, m, 32);
    return v;
}

// LayerNorm over one 96-channel row, one wave (lane handles 3 channels).
__device__ inline void ln_row(const float* __restrict__ src,
                              const float* __restrict__ g,
                              const float* __restrict__ b,
                              h16* __restrict__ dst, int lane) {
    float v0 = src[lane * 3 + 0], v1 = src[lane * 3 + 1], v2 = src[lane * 3 + 2];
    float mu = wave32_sum(v0 + v1 + v2) * (1.0f / 96.0f);
    float d0 = v0 - mu, d1 = v1 - mu, d2 = v2 - mu;
    float var = wave32_sum(d0 * d0 + d1 * d1 + d2 * d2) * (1.0f / 96.0f);
    float rs = rsqrtf(var + EPS_);
    dst[lane * 3 + 0] = (h16)(d0 * rs * g[lane * 3 + 0] + b[lane * 3 + 0]);
    dst[lane * 3 + 1] = (h16)(d1 * rs * g[lane * 3 + 1] + b[lane * 3 + 1]);
    dst[lane * 3 + 2] = (h16)(d2 * rs * g[lane * 3 + 2] + b[lane * 3 + 2]);
}

// ===========================================================================
// kernels
// ===========================================================================

__global__ void k_wconv(const float* __restrict__ qkvw, const float* __restrict__ projw,
                        const float* __restrict__ fc1w, const float* __restrict__ fc2w,
                        h16* __restrict__ wt) {
    int i = blockIdx.x * 256 + threadIdx.x;
    if (i < 27648) wt[WQKV + i]  = (h16)qkvw[i];
    if (i < 9216)  wt[WPROJ + i] = (h16)projw[i];
    if (i < 36864) { wt[WFC1 + i] = (h16)fc1w[i]; wt[WFC2 + i] = (h16)fc2w[i]; }
}

// LN1 + cyclic shift (-2 each dim) + 4D window partition -> f16 windows.
__global__ void k_ln1(const float* __restrict__ x, const float* __restrict__ g,
                      const float* __restrict__ b, h16* __restrict__ win) {
    int lane = threadIdx.x & 31;
    int tok = blockIdx.x * 8 + (threadIdx.x >> 5);      // window-major token id
    int w = tok >> 8, n = tok & 255;
    int bb = w >> 8, wb = w & 255;
    int dW = (wb >> 6) & 3, hW = (wb >> 4) & 3, wW = (wb >> 2) & 3, tW = wb & 3;
    int di = (n >> 6) & 3, hh = (n >> 4) & 3, wi = (n >> 2) & 3, ti = n & 3;
    int sd = (dW * 4 + di + 2) & 15, sh = (hW * 4 + hh + 2) & 15;
    int sw = (wW * 4 + wi + 2) & 15, st = (tW * 4 + ti + 2) & 15;
    const float* src = x + (((((size_t)bb * 16 + sd) * 16 + sh) * 16 + sw) * 16 + st) * 96;
    ln_row(src, g, b, win + (size_t)tok * 96, lane);
}

// QKV GEMM per window: [256x96] @ [96x288] ; scatter into head-major q/k
// (token-major rows of 32, hd padded) and vT (feature-major rows of 256).
// Pad elements pre-zeroed by memset. q scaled by hd^-0.5.
__global__ void k_qkv(const h16* __restrict__ win, const h16* __restrict__ wt,
                      const float* __restrict__ qkvb,
                      h16* __restrict__ q, h16* __restrict__ k, h16* __restrict__ vT) {
    int lane = threadIdx.x & 31, hi = lane >> 4, ln = lane & 15;
    int tid = blockIdx.x * 8 + (threadIdx.x >> 5);
    int w = tid / 288, rem = tid % 288, mt = rem / 18, nt = rem % 18;
    const h16* A0 = win + (size_t)w * 256 * 96 + (size_t)mt * 16 * 96;
    const h16* B0 = wt + WQKV + (size_t)nt * 16 * 96;
    v8f acc = {0.f, 0.f, 0.f, 0.f, 0.f, 0.f, 0.f, 0.f};
#pragma unroll
    for (int k0 = 0; k0 < 96; k0 += 32)
        acc = WMMA(load_a_frag(A0 + k0, 96, lane), load_b_frag_rm(B0 + k0, 96, lane), acc);
    int col = nt * 16 + ln;
    int t = col / 96, cc = col % 96, hd = cc / 24, e = cc % 24;
    float bias = qkvb[col];
    if (t == 2) {
        // vT[win][head][e][token] -> contiguous along token
        h16* dst = vT + (((size_t)w * 4 + hd) * 32 + e) * 256;
#pragma unroll
        for (int r = 0; r < 8; ++r) {
            int m = mt * 16 + r + hi * 8;
            dst[m] = (h16)(acc[r] + bias);
        }
    } else {
        h16* dst = (t == 0) ? q : k;
        float sc = (t == 0) ? SCALE_ : 1.0f;
#pragma unroll
        for (int r = 0; r < 8; ++r) {
            int m = mt * 16 + r + hi * 8;
            dst[(((size_t)w * 4 + hd) * 256 + m) * 32 + e] = (h16)((acc[r] + bias) * sc);
        }
    }
}

// Attention: one workgroup per (window, head). Each wave handles two 16-row
// blocks sequentially: S = q@k^T (16 WMMAs, K=32 padded), +mask, f32 softmax
// via 16-lane shuffles, P->LDS (f16), then out = P@v (16 WMMAs, vT fragments
// are two 16-byte lane loads each).
__global__ void k_attn(const h16* __restrict__ q, const h16* __restrict__ k,
                       const h16* __restrict__ vT, const float* __restrict__ mask,
                       h16* __restrict__ ao) {
    extern __shared__ h16 Plds[];                      // 8 waves * 16*256 halves
    int wv = threadIdx.x >> 5, lane = threadIdx.x & 31, hi = lane >> 4, ln = lane & 15;
    int win = blockIdx.x >> 2, head = blockIdx.x & 3;
    const h16* qb = q + ((size_t)win * 4 + head) * 256 * 32;
    const h16* kb = k + ((size_t)win * 4 + head) * 256 * 32;
    const h16* vb = vT + ((size_t)win * 4 + head) * 32 * 256;
    const float* mw = mask + (size_t)(win & 255) * 65536;
    h16* P = Plds + (size_t)wv * 4096;
    const v8f z = {0.f, 0.f, 0.f, 0.f, 0.f, 0.f, 0.f, 0.f};

    for (int it = 0; it < 2; ++it) {
        int m0 = (wv * 2 + it) * 16;
        v8f acc[16];
#pragma unroll
        for (int t = 0; t < 16; ++t) acc[t] = z;
        v16h aq = load_a_frag(qb + (size_t)m0 * 32, 32, lane);
#pragma unroll
        for (int t = 0; t < 16; ++t)
            acc[t] = WMMA(aq, load_b_frag_rm(kb + (size_t)t * 16 * 32, 32, lane), acc[t]);

        // mask add + row softmax (rows r+hi*8 live in this lane across 16 tiles)
#pragma unroll
        for (int r = 0; r < 8; ++r) {
            int m = m0 + r + hi * 8;
            const float* mrow = mw + (size_t)m * 256 + ln;
            float rmax = -3.0e38f;
#pragma unroll
            for (int t = 0; t < 16; ++t) {
                acc[t][r] += mrow[t * 16];
                rmax = fmaxf(rmax, acc[t][r]);
            }
#pragma unroll
            for (int msk = 1; msk < 16; msk <<= 1)
                rmax = fmaxf(rmax, __shfl_xor(rmax, msk, 32));
            float rsum = 0.f;
#pragma unroll
            for (int t = 0; t < 16; ++t) {
                float e = __expf(acc[t][r] - rmax);
                acc[t][r] = e;
                rsum += e;
            }
#pragma unroll
            for (int msk = 1; msk < 16; msk <<= 1)
                rsum += __shfl_xor(rsum, msk, 32);
            float inv = 1.0f / rsum;
#pragma unroll
            for (int t = 0; t < 16; ++t)
                P[(size_t)(r + hi * 8) * 256 + t * 16 + ln] = (h16)(acc[t][r] * inv);
        }

        // out = P @ v : B(k=token, n=e) = vT[n][k] -> row-major fragment loads
        v8f o0 = z, o1 = z;
#pragma unroll
        for (int k0 = 0; k0 < 256; k0 += 32) {
            v16h ap = load_a_frag(P + k0, 256, lane);
            o0 = WMMA(ap, load_b_frag_rm(vb + k0, 256, lane), o0);
            o1 = WMMA(ap, load_b_frag_rm(vb + (size_t)16 * 256 + k0, 256, lane), o1);
        }
#pragma unroll
        for (int r = 0; r < 8; ++r) {
            int m = m0 + r + hi * 8;
            h16* orow = ao + ((size_t)win * 256 + m) * 96 + head * 24;
            orow[ln] = (h16)o0[r];
            if (ln < 8) orow[16 + ln] = (h16)o1[r];
        }
    }
}

// proj GEMM + window reverse + reverse shift + residual: x2 = x + scatter(out)
__global__ void k_proj(const h16* __restrict__ ao, const h16* __restrict__ wt,
                       const float* __restrict__ projb, const float* __restrict__ x,
                       float* __restrict__ x2) {
    int lane = threadIdx.x & 31, hi = lane >> 4, ln = lane & 15;
    int tid = blockIdx.x * 8 + (threadIdx.x >> 5);
    int w = tid / 96, rem = tid % 96, mt = rem / 6, nt = rem % 6;
    const h16* A0 = ao + (size_t)w * 256 * 96 + (size_t)mt * 16 * 96;
    const h16* B0 = wt + WPROJ + (size_t)nt * 16 * 96;
    v8f acc = {0.f, 0.f, 0.f, 0.f, 0.f, 0.f, 0.f, 0.f};
#pragma unroll
    for (int k0 = 0; k0 < 96; k0 += 32)
        acc = WMMA(load_a_frag(A0 + k0, 96, lane), load_b_frag_rm(B0 + k0, 96, lane), acc);
    int col = nt * 16 + ln;
    float bias = projb[col];
    int bb = w >> 8, wb = w & 255;
    int dW = (wb >> 6) & 3, hW = (wb >> 4) & 3, wW = (wb >> 2) & 3, tW = wb & 3;
#pragma unroll
    for (int r = 0; r < 8; ++r) {
        int m = mt * 16 + r + hi * 8;
        int di = (m >> 6) & 3, hh = (m >> 4) & 3, wi = (m >> 2) & 3, ti = m & 3;
        int d = (dW * 4 + di + 2) & 15, h = (hW * 4 + hh + 2) & 15;
        int ww = (wW * 4 + wi + 2) & 15, t = (tW * 4 + ti + 2) & 15;
        size_t idx = (((((size_t)bb * 16 + d) * 16 + h) * 16 + ww) * 16 + t) * 96 + col;
        x2[idx] = x[idx] + acc[r] + bias;
    }
}

__global__ void k_ln2(const float* __restrict__ x2, const float* __restrict__ g,
                      const float* __restrict__ b, h16* __restrict__ h) {
    int lane = threadIdx.x & 31;
    int tok = blockIdx.x * 8 + (threadIdx.x >> 5);
    ln_row(x2 + (size_t)tok * 96, g, b, h + (size_t)tok * 96, lane);
}

// FC1 + exact-erf GELU: [131072 x 96] @ [96 x 384]
__global__ void k_fc1(const h16* __restrict__ h, const h16* __restrict__ wt,
                      const float* __restrict__ fc1b, h16* __restrict__ h2) {
    int lane = threadIdx.x & 31, hi = lane >> 4, ln = lane & 15;
    int tid = blockIdx.x * 8 + (threadIdx.x >> 5);
    int mt = tid / 24, nt = tid % 24;
    const h16* A0 = h + (size_t)mt * 16 * 96;
    const h16* B0 = wt + WFC1 + (size_t)nt * 16 * 96;
    v8f acc = {0.f, 0.f, 0.f, 0.f, 0.f, 0.f, 0.f, 0.f};
#pragma unroll
    for (int k0 = 0; k0 < 96; k0 += 32)
        acc = WMMA(load_a_frag(A0 + k0, 96, lane), load_b_frag_rm(B0 + k0, 96, lane), acc);
    int col = nt * 16 + ln;
    float bias = fc1b[col];
#pragma unroll
    for (int r = 0; r < 8; ++r) {
        int tok = mt * 16 + r + hi * 8;
        float vv = acc[r] + bias;
        float ge = 0.5f * vv * (1.0f + erff(vv * 0.70710678118654752f));
        h2[(size_t)tok * 384 + col] = (h16)ge;
    }
}

// FC2 + residual: out = x2 + [131072 x 384] @ [384 x 96]
__global__ void k_fc2(const h16* __restrict__ h2, const h16* __restrict__ wt,
                      const float* __restrict__ fc2b, const float* __restrict__ x2,
                      float* __restrict__ out) {
    int lane = threadIdx.x & 31, hi = lane >> 4, ln = lane & 15;
    int tid = blockIdx.x * 8 + (threadIdx.x >> 5);
    int mt = tid / 6, nt = tid % 6;
    const h16* A0 = h2 + (size_t)mt * 16 * 384;
    const h16* B0 = wt + WFC2 + (size_t)nt * 16 * 384;
    v8f acc = {0.f, 0.f, 0.f, 0.f, 0.f, 0.f, 0.f, 0.f};
#pragma unroll
    for (int k0 = 0; k0 < 384; k0 += 32)
        acc = WMMA(load_a_frag(A0 + k0, 384, lane), load_b_frag_rm(B0 + k0, 384, lane), acc);
    int col = nt * 16 + ln;
    float bias = fc2b[col];
#pragma unroll
    for (int r = 0; r < 8; ++r) {
        int tok = mt * 16 + r + hi * 8;
        size_t idx = (size_t)tok * 96 + col;
        out[idx] = x2[idx] + acc[r] + bias;
    }
}

// ===========================================================================
// launcher
// ===========================================================================

extern "C" void kernel_launch(void* const* d_in, const int* in_sizes, int n_in,
                              void* d_out, int out_size, void* d_ws, size_t ws_size,
                              hipStream_t stream) {
    const float* x     = (const float*)d_in[0];
    const float* mask  = (const float*)d_in[1];
    const float* n1g   = (const float*)d_in[2];
    const float* n1b   = (const float*)d_in[3];
    const float* qkvw  = (const float*)d_in[4];
    const float* qkvb  = (const float*)d_in[5];
    const float* projw = (const float*)d_in[6];
    const float* projb = (const float*)d_in[7];
    const float* n2g   = (const float*)d_in[8];
    const float* n2b   = (const float*)d_in[9];
    const float* fc1w  = (const float*)d_in[10];
    const float* fc1b  = (const float*)d_in[11];
    const float* fc2w  = (const float*)d_in[12];
    const float* fc2b  = (const float*)d_in[13];

    char* ws = (char*)d_ws;
    h16*   winb = (h16*)(ws + OFF_WIN);
    h16*   q    = (h16*)(ws + OFF_Q);
    h16*   k    = (h16*)(ws + OFF_K);
    h16*   vT   = (h16*)(ws + OFF_V);
    h16*   ao   = (h16*)(ws + OFF_AO);
    float* x2   = (float*)(ws + OFF_X2);
    h16*   hbuf = (h16*)(ws + OFF_H);
    h16*   h2   = (h16*)(ws + OFF_H2);
    h16*   wt   = (h16*)(ws + OFF_WT);
    float* out  = (float*)d_out;

    k_wconv<<<144, 256, 0, stream>>>(qkvw, projw, fc1w, fc2w, wt);
    hipMemsetAsync(ws + OFF_Q, 0, 3ull * 33554432ull, stream);   // zero q/k/vT pads
    k_ln1 <<<16384, 256, 0, stream>>>(x, n1g, n1b, winb);
    k_qkv <<<18432, 256, 0, stream>>>(winb, wt, qkvb, q, k, vT);
    k_attn<<< 2048, 256, 65536, stream>>>(q, k, vT, mask, ao);
    k_proj<<< 6144, 256, 0, stream>>>(ao, wt, projb, x, x2);
    k_ln2 <<<16384, 256, 0, stream>>>(x2, n2g, n2b, hbuf);
    k_fc1 <<<24576, 256, 0, stream>>>(hbuf, wt, fc1b, h2);
    k_fc2 <<< 6144, 256, 0, stream>>>(h2, wt, fc2b, x2, out);
}
